// GATLayer_29695403884564
// MI455X (gfx1250) — matrix operand order repaired and checked by reading, not compiled, not run on hardware
//
#include <hip/hip_runtime.h>
#include <hip/hip_bf16.h>

typedef __attribute__((ext_vector_type(16))) _Float16 v16h;
typedef __attribute__((ext_vector_type(8)))  float    v8f;
typedef __attribute__((ext_vector_type(4)))  int      v4i;

#define Nn    8192
#define INC   512
#define OUTC  64
#define ALPHA 0.2f

// ---------------------------------------------------------------------------
// prep: h = x @ W  (f32, row-major 8192x64)  +  hbT (f16, transposed 64x8192)
// block = 256 threads = 4 rows x 64 cols, grid = 2048
// ---------------------------------------------------------------------------
__global__ void gat_prep(const float* __restrict__ x, const float* __restrict__ W,
                         float* __restrict__ h, _Float16* __restrict__ hbT) {
    int tid = threadIdx.x;
    int c   = tid & 63;
    int row = blockIdx.x * 4 + (tid >> 6);
    const float4* xr = reinterpret_cast<const float4*>(x + (size_t)row * INC);
    float sum = 0.f;
#pragma unroll 4
    for (int kk = 0; kk < INC / 4; ++kk) {
        float4 xv = xr[kk];
        int k = kk * 4;
        sum += xv.x * W[(k + 0) * OUTC + c];
        sum += xv.y * W[(k + 1) * OUTC + c];
        sum += xv.z * W[(k + 2) * OUTC + c];
        sum += xv.w * W[(k + 3) * OUTC + c];
    }
    h[(size_t)row * OUTC + c] = sum;
    hbT[(size_t)c * Nn + row] = (_Float16)sum;
}

// ---------------------------------------------------------------------------
// score: s_src = h @ a[:64], s_dst = h @ a[64:]
// ---------------------------------------------------------------------------
__global__ void gat_score(const float* __restrict__ h, const float* __restrict__ a,
                          float* __restrict__ s_src, float* __restrict__ s_dst) {
    int row = blockIdx.x * blockDim.x + threadIdx.x;
    float ss = 0.f, sd = 0.f;
#pragma unroll 8
    for (int c = 0; c < OUTC; ++c) {
        float hv = h[(size_t)row * OUTC + c];
        ss += hv * a[c];
        sd += hv * a[OUTC + c];
    }
    s_src[row] = ss;
    s_dst[row] = sd;
}

// ---------------------------------------------------------------------------
// reduce: sdmax = max_j s_dst[j]  (single block)
// ---------------------------------------------------------------------------
__global__ void gat_reduce(const float* __restrict__ s_dst, float* __restrict__ sdmax) {
    __shared__ float sm[256];
    int tid = threadIdx.x;
    float m = -3.4e38f;
    for (int i = tid; i < Nn; i += 256) m = fmaxf(m, s_dst[i]);
    sm[tid] = m;
    __syncthreads();
    for (int s = 128; s > 0; s >>= 1) {
        if (tid < s) sm[tid] = fmaxf(sm[tid], sm[tid + s]);
        __syncthreads();
    }
    if (tid == 0) sdmax[0] = sm[0];
}

// ---------------------------------------------------------------------------
// attn: streaming masked softmax + P@H via v_wmma_f32_16x16x32_f16
// grid = 512 blocks (16-row tiles), block = 256 threads = 8 waves,
// wave w owns K range [w*1024, (w+1)*1024)
// amdgpu_waves_per_eu(4): relax VGPR budget so all 4 B operands stay live
// ---------------------------------------------------------------------------
__global__ void __launch_bounds__(256)
__attribute__((amdgpu_waves_per_eu(4)))
gat_attn(const int* __restrict__ adj,
         const float* __restrict__ s_src,
         const float* __restrict__ s_dst,
         const float* __restrict__ sdmax,
         const _Float16* __restrict__ hbT,
         float* __restrict__ out) {
    __shared__ float sh[Nn];      // phase 1: s_dst stage; phase 2: per-wave acc
    __shared__ float lbuf[8 * 16];

    const int tid  = threadIdx.x;
    const int lane = tid & 31;
    const int wave = tid >> 5;
    const int r    = lane & 15;
    const int row0 = blockIdx.x * 16;
    const int grow = row0 + r;

    // stage s_dst into LDS (vectorized)
    {
        const float4* g = reinterpret_cast<const float4*>(s_dst);
        float4* s = reinterpret_cast<float4*>(sh);
        for (int i = tid; i < Nn / 4; i += 256) s[i] = g[i];
    }
    __syncthreads();

    const float ssr = s_src[grow];
    float bnd = ssr + sdmax[0];
    const float shift = bnd > 0.f ? bnd : ALPHA * bnd;   // per-row score upper bound

    const int klo = (lane < 16) ? 0 : 8;        // A-matrix K ownership (16-bit layout)
    const int bkh = (lane < 16) ? 0 : 8;        // B-matrix K ownership (same as A)
    const int ncol = lane & 15;

    const int* __restrict__ arow = adj + (size_t)grow * Nn;
    const int kbase = wave * (Nn / 8);

    v8f acc0 = {}, acc1 = {}, acc2 = {}, acc3 = {};
    float ls0 = 0.f, ls1 = 0.f, ls2 = 0.f, ls3 = 0.f;   // split row-sum chains

    for (int kk = 0; kk < Nn / 8; kk += 32) {
        const int j0 = kbase + kk;
        __builtin_prefetch(arow + j0 + 256, 0, 1);   // stream adjacency ahead

        // ---- adjacency: 16 ints, non-temporal (one-shot 256MB stream) ----
        const v4i* pa = reinterpret_cast<const v4i*>(arow + j0 + klo);
        const v4i* pb = reinterpret_cast<const v4i*>(arow + j0 + 16 + klo);
        v4i a0 = __builtin_nontemporal_load(pa);
        v4i a1 = __builtin_nontemporal_load(pa + 1);
        v4i b0 = __builtin_nontemporal_load(pb);
        v4i b1 = __builtin_nontemporal_load(pb + 1);
        int am[16] = {a0[0], a0[1], a0[2], a0[3], a1[0], a1[1], a1[2], a1[3],
                      b0[0], b0[1], b0[2], b0[3], b1[0], b1[1], b1[2], b1[3]};

        // ---- s_dst values (LDS) for same K pattern ----
        const float4* qa = reinterpret_cast<const float4*>(&sh[j0 + klo]);
        const float4* qb = reinterpret_cast<const float4*>(&sh[j0 + 16 + klo]);
        float4 f0 = qa[0], f1 = qa[1], g0 = qb[0], g1 = qb[1];
        float sd[16] = {f0.x, f0.y, f0.z, f0.w, f1.x, f1.y, f1.z, f1.w,
                        g0.x, g0.y, g0.z, g0.w, g1.x, g1.y, g1.z, g1.w};

        // ---- masked leaky-relu scores -> exp -> f16 A operand (branchless) ----
        v16h av;
        float e[16];
#pragma unroll
        for (int i = 0; i < 16; ++i) {
            float sc = ssr + sd[i];
            sc = sc > 0.f ? sc : ALPHA * sc;
            float msk = (am[i] > 0) ? 1.0f : 0.0f;       // cndmask, no branch
            e[i] = msk * __expf(sc - shift);             // exp always executed
            av[i] = (_Float16)e[i];
        }
#pragma unroll
        for (int i = 0; i < 4; ++i) {                    // 4 parallel sum chains
            ls0 += e[i];
            ls1 += e[4 + i];
            ls2 += e[8 + i];
            ls3 += e[12 + i];
        }

        // ---- B operands: hbT[N][j0+bkh .. +7] and hbT[N][j0+16+bkh .. +7] ----
        union { v16h v; uint4 q[2]; } B0, B1, B2, B3;
        {
            const uint4* p0 = reinterpret_cast<const uint4*>(hbT + (size_t)(ncol)      * Nn + j0 + bkh);
            const uint4* p1 = reinterpret_cast<const uint4*>(hbT + (size_t)(ncol + 16) * Nn + j0 + bkh);
            const uint4* p2 = reinterpret_cast<const uint4*>(hbT + (size_t)(ncol + 32) * Nn + j0 + bkh);
            const uint4* p3 = reinterpret_cast<const uint4*>(hbT + (size_t)(ncol + 48) * Nn + j0 + bkh);
            B0.q[0] = p0[0]; B0.q[1] = p0[2];
            B1.q[0] = p1[0]; B1.q[1] = p1[2];
            B2.q[0] = p2[0]; B2.q[1] = p2[2];
            B3.q[0] = p3[0]; B3.q[1] = p3[2];
        }

        acc0 = __builtin_amdgcn_wmma_f32_16x16x32_f16(false, av, false, B0.v, (short)0, acc0, false, false);
        acc1 = __builtin_amdgcn_wmma_f32_16x16x32_f16(false, av, false, B1.v, (short)0, acc1, false, false);
        acc2 = __builtin_amdgcn_wmma_f32_16x16x32_f16(false, av, false, B2.v, (short)0, acc2, false, false);
        acc3 = __builtin_amdgcn_wmma_f32_16x16x32_f16(false, av, false, B3.v, (short)0, acc3, false, false);
    }

    float lrow = (ls0 + ls1) + (ls2 + ls3);
    // combine partial row sums across the two lane halves (same row r)
    lrow += __shfl_xor(lrow, 16);

    // ---- cross-wave combine: plain sums (shared shift => no max merge) ----
    __syncthreads();                       // all waves done reading s_dst from sh
    {
        int mbase = (lane < 16) ? 0 : 8;   // C layout: VGPR v -> M = v + mbase
#pragma unroll
        for (int v = 0; v < 8; ++v) {
            int M = v + mbase;
            sh[wave * 1024 + M * 64 + (ncol +  0)] = acc0[v];
            sh[wave * 1024 + M * 64 + (ncol + 16)] = acc1[v];
            sh[wave * 1024 + M * 64 + (ncol + 32)] = acc2[v];
            sh[wave * 1024 + M * 64 + (ncol + 48)] = acc3[v];
        }
        if (lane < 16) lbuf[wave * 16 + r] = lrow;
    }
    __syncthreads();

    // ---- finalize: divide by row sum, ELU, store ----
    for (int e2 = tid; e2 < 16 * 64; e2 += 256) {
        int M = e2 >> 6;
        float s = 0.f, l = 0.f;
#pragma unroll
        for (int w = 0; w < 8; ++w) {
            s += sh[w * 1024 + e2];
            l += lbuf[w * 16 + M];
        }
        float val = s / l;
        out[(size_t)(row0 + M) * OUTC + (e2 & 63)] = val > 0.f ? val : expm1f(val);
    }
}

// ---------------------------------------------------------------------------
extern "C" void kernel_launch(void* const* d_in, const int* in_sizes, int n_in,
                              void* d_out, int out_size, void* d_ws, size_t ws_size,
                              hipStream_t stream) {
    const float* x   = (const float*)d_in[0];
    const int*   adj = (const int*)  d_in[1];
    const float* W   = (const float*)d_in[2];
    const float* a   = (const float*)d_in[3];
    float*       out = (float*)d_out;

    char* ws = (char*)d_ws;
    float*     h     = (float*)    (ws);                            // 2 MB
    _Float16*  hbT   = (_Float16*) (ws + (size_t)2 * 1024 * 1024);  // 1 MB
    float*     s_src = (float*)    (ws + (size_t)3 * 1024 * 1024);  // 32 KB
    float*     s_dst = (float*)    (ws + (size_t)3 * 1024 * 1024 + 32 * 1024);
    float*     sdmax = (float*)    (ws + (size_t)3 * 1024 * 1024 + 64 * 1024);

    gat_prep  <<<Nn / 4, 256, 0, stream>>>(x, W, h, hbT);
    gat_score <<<Nn / 256, 256, 0, stream>>>(h, a, s_src, s_dst);
    gat_reduce<<<1, 256, 0, stream>>>(s_dst, sdmax);
    gat_attn  <<<Nn / 16, 256, 0, stream>>>(adj, s_src, s_dst, sdmax, hbT, out);
}
